// AttnDecoderRNN_47639777247738
// MI455X (gfx1250) — matrix operand order repaired and checked by reading, not compiled, not run on hardware
//
#include <hip/hip_runtime.h>
#include <hip/hip_bf16.h>

// AttnDecoderRNN step for MI455X (gfx1250, wave32).
// bf16 v_wmma_f32_16x16x32_bf16 for the two big GEMMs (combine 8.6 GF, gates 68.7 GF),
// TDM tensor_load_to_lds for A-tile staging (hardware row padding, TENSORcnt),
// fp32 VALU for attention softmax / LSTM nonlinearity / output projection.

#define BATCH 16384
#define HDIM  512
#define SDIM  6

typedef __attribute__((ext_vector_type(16))) __bf16 v16bf;
typedef __attribute__((ext_vector_type(8)))  float  v8f;
typedef __attribute__((ext_vector_type(4)))  unsigned int u32x4;
typedef __attribute__((ext_vector_type(8)))  int i32x8;
typedef __attribute__((ext_vector_type(4)))  int i32x4;

union Frag16 { uint4 u[2]; v16bf v; };

__device__ __forceinline__ unsigned short f32_to_bf16_bits(float f) {
  unsigned int u = __float_as_uint(f);
  u += 0x7FFFu + ((u >> 16) & 1u);      // round-to-nearest-even
  return (unsigned short)(u >> 16);
}

// ---------------------------------------------------------------------------
// TDM: async-DMA a 64-row x 512-half bf16 tile (contiguous rows) into LDS,
// padding 4 DWORDs after every 256 DWORDs -> LDS row stride 260 DWORDs
// (= 65 uint4), matching the bank-conflict-free A-fragment layout.
// D# fields per CDNA5 ISA.
// ---------------------------------------------------------------------------
__device__ __forceinline__ void tdm_load_tile_64x512(const unsigned short* gsrc,
                                                     unsigned lds_byte_off) {
  unsigned long long ga = (unsigned long long)gsrc;
  u32x4 g0;
  g0[0] = 1u;                                              // count=1, user desc
  g0[1] = lds_byte_off;                                    // lds_addr
  g0[2] = (unsigned)(ga & 0xFFFFFFFFu);                    // global_addr[31:0]
  g0[3] = (unsigned)((ga >> 32) & 0x01FFFFFFu) | (2u << 30); // addr[56:32] | type=2
  i32x8 g1;
  g1[0] = (int)((1u << 16)      // data_size = 2 bytes
              | (1u << 20)      // pad_enable
              | (7u << 22)      // pad_interval: 256 DWORDs
              | (3u << 25));    // pad_amount: 4 DWORDs
  g1[1] = (int)(512u << 16);    // tensor_dim0 = 512 (abar_addr = 0)
  g1[2] = (int)(64u << 16);     // tensor_dim0 hi | tensor_dim1 = 64
  g1[3] = (int)(512u << 16);    // tensor_dim1 hi | tile_dim0 = 512
  g1[4] = 64;                   // tile_dim1 = 64, tile_dim2 = 0
  g1[5] = 512;                  // tensor_dim0_stride = 512
  g1[6] = 0;
  g1[7] = 0;
  i32x4 z4 = {0, 0, 0, 0};
#if defined(__clang_major__) && (__clang_major__ >= 23)
  i32x8 z8 = {0, 0, 0, 0, 0, 0, 0, 0};
  __builtin_amdgcn_tensor_load_to_lds(g0, g1, z4, z4, z8, 0);
#else
  __builtin_amdgcn_tensor_load_to_lds(g0, g1, z4, z4, 0);
#endif
}

// ---------------------------------------------------------------------------
// Kernel 1: convert W_ih, W_hh, W_comb[:,1:], hidden to bf16 in workspace.
// ws layout (u16 elems): [0,NW) Wih16 | [NW,2NW) Whh16 | [2NW,2NW+NC) Wc16 |
//                        [2NW+NC, +NH) hidden16  (then attnapp16, x16 follow)
// ---------------------------------------------------------------------------
__global__ void prep_kernel(const float* __restrict__ Wih,
                            const float* __restrict__ Whh,
                            const float* __restrict__ Wcomb,
                            const float* __restrict__ hidden,
                            unsigned short* __restrict__ ws16) {
  const int NW = 2048 * 512;
  const int NC = 512 * 512;
  const int NH = BATCH * HDIM;
  const int total = NW + NW + NC + NH;
  for (int idx = blockIdx.x * blockDim.x + threadIdx.x; idx < total;
       idx += gridDim.x * blockDim.x) {
    float v;
    if (idx < NW)                v = Wih[idx];
    else if (idx < 2 * NW)       v = Whh[idx - NW];
    else if (idx < 2 * NW + NC) { int j = idx - 2 * NW;
                                  v = Wcomb[(size_t)(j >> 9) * 513 + 1 + (j & 511)]; }
    else                         v = hidden[idx - (2 * NW + NC)];
    ws16[idx] = f32_to_bf16_bits(v);
  }
}

// ---------------------------------------------------------------------------
// Kernel 2a: attn logits + softmax. One thread per row. W_attn accesses use
// uniform indices -> scalar loads via constant cache; hidden/cell via float4.
// ---------------------------------------------------------------------------
__global__ void attn_logits_kernel(const float* __restrict__ input,
                                   const float* __restrict__ hidden,
                                   const float* __restrict__ cell,
                                   const float* __restrict__ W_attn,
                                   const float* __restrict__ b_attn,
                                   float* __restrict__ out_w) {
  int b = blockIdx.x * blockDim.x + threadIdx.x;
  float z[SDIM];
  float xin = input[b];
#pragma unroll
  for (int s = 0; s < SDIM; ++s) z[s] = b_attn[s] + xin * W_attn[s * 1025];
  const float* hrow = hidden + (size_t)b * HDIM;
  const float* crow = cell   + (size_t)b * HDIM;
  for (int k = 0; k < HDIM; k += 4) {
    float4 hv = *(const float4*)(hrow + k);
    float4 cv = *(const float4*)(crow + k);
#pragma unroll
    for (int s = 0; s < SDIM; ++s) {
      const float* wa = W_attn + s * 1025;
      z[s] += hv.x * wa[1 + k]   + hv.y * wa[2 + k]
            + hv.z * wa[3 + k]   + hv.w * wa[4 + k]
            + cv.x * wa[513 + k] + cv.y * wa[514 + k]
            + cv.z * wa[515 + k] + cv.w * wa[516 + k];
    }
  }
  float m = z[0];
#pragma unroll
  for (int s = 1; s < SDIM; ++s) m = fmaxf(m, z[s]);
  float e[SDIM], sum = 0.f;
#pragma unroll
  for (int s = 0; s < SDIM; ++s) { e[s] = __expf(z[s] - m); sum += e[s]; }
  float inv = 1.f / sum;
#pragma unroll
  for (int s = 0; s < SDIM; ++s) out_w[(size_t)b * SDIM + s] = e[s] * inv;
}

// ---------------------------------------------------------------------------
// Kernel 2b: attn_applied[b,k] = sum_s w[b,s] * enc[b,s,k]  (coalesced 201 MB
// pass), output bf16. 64 rows per block.
// ---------------------------------------------------------------------------
__global__ void attn_apply_kernel(const float* __restrict__ enc,
                                  const float* __restrict__ attn_w,
                                  unsigned short* __restrict__ app16) {
  __shared__ float w[64][SDIM];
  int b0 = blockIdx.x * 64;
  for (int i = threadIdx.x; i < 64 * SDIM; i += 256)
    w[i / SDIM][i % SDIM] = attn_w[(size_t)(b0 + i / SDIM) * SDIM + i % SDIM];
  __syncthreads();
  for (int e = threadIdx.x; e < 64 * HDIM; e += 256) {
    int r = e >> 9, k = e & 511;
    const float* er = enc + ((size_t)(b0 + r)) * SDIM * HDIM + k;
    float acc = 0.f;
#pragma unroll
    for (int s = 0; s < SDIM; ++s) acc += w[r][s] * er[(size_t)s * HDIM];
    app16[(size_t)(b0 + r) * HDIM + k] = f32_to_bf16_bits(acc);
  }
}

// ---------------------------------------------------------------------------
// Kernel 3: x = relu(attnapp @ Wc^T + Wcomb[:,0]*input + b_comb), bf16 out.
// Block: 64 batch rows x 32 cols, 8 waves, 1 WMMA tile per wave.
// A tile staged via TDM with hardware padding to 65-uint4 row stride.
// ---------------------------------------------------------------------------
__global__ void combine_kernel(const unsigned short* __restrict__ app16,
                               const unsigned short* __restrict__ Wc16,
                               const float* __restrict__ Wcomb,
                               const float* __restrict__ b_comb,
                               const float* __restrict__ input,
                               unsigned short* __restrict__ x16) {
  extern __shared__ uint4 ldsA[];                 // 64 * 65 uint4
  int b0 = blockIdx.x * 64;
  if (threadIdx.x < 32) {
    tdm_load_tile_64x512(app16 + (size_t)b0 * HDIM, (unsigned)(size_t)&ldsA[0]);
    __builtin_amdgcn_s_wait_tensorcnt(0);
  }
  __syncthreads();

  int wid  = threadIdx.x >> 5;
  int lane = threadIdx.x & 31;
  int mtile = wid & 3;                  // 4 row tiles
  int nt    = wid >> 2;                 // 2 col tiles
  int n0 = blockIdx.y * 32 + nt * 16;
  int g = lane >> 4, m = lane & 15;
  int arow = mtile * 16 + m;
  const uint4* gB = (const uint4*)Wc16;

  v8f acc = {};
  for (int k0 = 0; k0 < HDIM; k0 += 32) {
    Frag16 a, bfr;
    int ai = arow * 65 + (k0 >> 3) + 2 * g;       // halves k0+16g .. +7
    a.u[0] = ldsA[ai];
    a.u[1] = ldsA[ai + 1];                        // halves k0+16g+8 .. +15
    size_t bi = (((size_t)(n0 + m)) * HDIM + (size_t)k0 + 16u * g) >> 3;
    bfr.u[0] = gB[bi];
    bfr.u[1] = gB[bi + 1];
    acc = __builtin_amdgcn_wmma_f32_16x16x32_bf16(false, a.v, false, bfr.v,
                                                  (short)0, acc, false, false);
  }
  int n = n0 + m;
  float bias = b_comb[n];
  float w0   = Wcomb[(size_t)n * 513];
#pragma unroll
  for (int r = 0; r < 8; ++r) {
    int b = b0 + mtile * 16 + r + 8 * g;          // C/D layout: M = r + 8*(lane/16)
    float v = acc[r] + bias + w0 * input[b];
    v = fmaxf(v, 0.f);
    x16[(size_t)b * HDIM + n] = f32_to_bf16_bits(v);
  }
}

// ---------------------------------------------------------------------------
// Kernel 4: gates = x@Wih^T + hidden@Whh^T + b  -> fused LSTM elementwise.
// Each wave owns one 16x16 (batch, j) tile for ALL FOUR gates: A fragments
// are loaded once per k-step and feed 8 WMMAs (4 gates x 2 weight mats).
// Both A tiles (x, hidden) staged via TDM.
// ---------------------------------------------------------------------------
__global__ void gates_lstm_kernel(const unsigned short* __restrict__ x16,
                                  const unsigned short* __restrict__ h16,
                                  const unsigned short* __restrict__ Wih16,
                                  const unsigned short* __restrict__ Whh16,
                                  const float* __restrict__ b_ih,
                                  const float* __restrict__ b_hh,
                                  const float* __restrict__ cell,
                                  float* __restrict__ out_h,
                                  float* __restrict__ out_c) {
  extern __shared__ uint4 lds[];                  // X tile then H tile
  uint4* ldsX = lds;
  uint4* ldsH = lds + 64 * 65;
  int b0 = blockIdx.x * 64;
  if (threadIdx.x < 32) {
    tdm_load_tile_64x512(x16 + (size_t)b0 * HDIM, (unsigned)(size_t)&lds[0]);
    tdm_load_tile_64x512(h16 + (size_t)b0 * HDIM,
                         (unsigned)(size_t)&lds[0] + 64u * 65u * (unsigned)sizeof(uint4));
    __builtin_amdgcn_s_wait_tensorcnt(0);
  }
  __syncthreads();

  int wid  = threadIdx.x >> 5;
  int lane = threadIdx.x & 31;
  int mtile = wid & 3;
  int jt    = wid >> 2;                 // 0..1
  int j0 = blockIdx.y * 32 + jt * 16;
  int g = lane >> 4, m = lane & 15;
  int arow = mtile * 16 + m;
  const uint4* gBih = (const uint4*)Wih16;
  const uint4* gBhh = (const uint4*)Whh16;

  v8f acc[4] = {};                      // i, f, g, o accumulators
  for (int k0 = 0; k0 < HDIM; k0 += 32) {
    Frag16 ax, ah;
    int ai = arow * 65 + (k0 >> 3) + 2 * g;
    ax.u[0] = ldsX[ai]; ax.u[1] = ldsX[ai + 1];
    ah.u[0] = ldsH[ai]; ah.u[1] = ldsH[ai + 1];
#pragma unroll
    for (int gg = 0; gg < 4; ++gg) {
      size_t n  = (size_t)gg * HDIM + j0 + m;
      size_t bi = (n * HDIM + (size_t)k0 + 16u * g) >> 3;
      Frag16 bI, bH;
      bI.u[0] = gBih[bi]; bI.u[1] = gBih[bi + 1];
      bH.u[0] = gBhh[bi]; bH.u[1] = gBhh[bi + 1];
      acc[gg] = __builtin_amdgcn_wmma_f32_16x16x32_bf16(false, ax.v, false, bI.v,
                                                        (short)0, acc[gg], false, false);
      acc[gg] = __builtin_amdgcn_wmma_f32_16x16x32_bf16(false, ah.v, false, bH.v,
                                                        (short)0, acc[gg], false, false);
    }
  }

  int col = j0 + m;
  float bi_ = b_ih[col]            + b_hh[col];
  float bf_ = b_ih[HDIM + col]     + b_hh[HDIM + col];
  float bg_ = b_ih[2 * HDIM + col] + b_hh[2 * HDIM + col];
  float bo_ = b_ih[3 * HDIM + col] + b_hh[3 * HDIM + col];
#pragma unroll
  for (int r = 0; r < 8; ++r) {
    int b = b0 + mtile * 16 + r + 8 * g;
    size_t off = (size_t)b * HDIM + col;
    float iv = acc[0][r] + bi_;
    float fv = acc[1][r] + bf_;
    float gv = acc[2][r] + bg_;
    float ov = acc[3][r] + bo_;
    float si = 1.f / (1.f + __expf(-iv));
    float sf = 1.f / (1.f + __expf(-fv));
    float so = 1.f / (1.f + __expf(-ov));
    float cn = sf * cell[off] + si * tanhf(gv);
    float hn = so * tanhf(cn);
    out_c[off] = cn;
    out_h[off] = hn;
  }
}

// ---------------------------------------------------------------------------
// Kernel 5: output[b] = h_new[b,:] . W_out + b_out. One wave32 per row.
// ---------------------------------------------------------------------------
__global__ void out_proj_kernel(const float* __restrict__ h,
                                const float* __restrict__ W_out,
                                const float* __restrict__ b_out,
                                float* __restrict__ out) {
  int gtid = blockIdx.x * blockDim.x + threadIdx.x;
  int row  = gtid >> 5;
  int lane = gtid & 31;
  const float* hr = h + (size_t)row * HDIM;
  float p = 0.f;
  for (int j = lane; j < HDIM; j += 32) p += hr[j] * W_out[j];
#pragma unroll
  for (int off = 16; off > 0; off >>= 1) p += __shfl_down(p, off, 32);
  if (lane == 0) out[row] = p + b_out[0];
}

// ---------------------------------------------------------------------------
extern "C" void kernel_launch(void* const* d_in, const int* in_sizes, int n_in,
                              void* d_out, int out_size, void* d_ws, size_t ws_size,
                              hipStream_t stream) {
  (void)in_sizes; (void)n_in; (void)out_size; (void)ws_size;
  const float* input  = (const float*)d_in[0];
  const float* hidden = (const float*)d_in[1];
  const float* cell   = (const float*)d_in[2];
  const float* enc    = (const float*)d_in[3];
  const float* W_attn = (const float*)d_in[4];
  const float* b_attn = (const float*)d_in[5];
  const float* W_comb = (const float*)d_in[6];
  const float* b_comb = (const float*)d_in[7];
  const float* W_ih   = (const float*)d_in[8];
  const float* W_hh   = (const float*)d_in[9];
  const float* b_ih   = (const float*)d_in[10];
  const float* b_hh   = (const float*)d_in[11];
  const float* W_out  = (const float*)d_in[12];
  const float* b_out  = (const float*)d_in[13];

  float* out   = (float*)d_out;                             // [B]
  float* out_h = out + BATCH;                               // [B,512]
  float* out_c = out_h + (size_t)BATCH * HDIM;              // [B,512]
  float* out_w = out_c + (size_t)BATCH * HDIM;              // [B,1,6]

  unsigned short* ws = (unsigned short*)d_ws;
  const size_t NW = 2048 * 512, NC = 512 * 512, NH = (size_t)BATCH * HDIM;
  unsigned short* Wih16 = ws;
  unsigned short* Whh16 = ws + NW;
  unsigned short* Wc16  = ws + 2 * NW;
  unsigned short* H16   = ws + 2 * NW + NC;
  unsigned short* A16   = H16 + NH;
  unsigned short* X16   = A16 + NH;

  prep_kernel<<<4096, 256, 0, stream>>>(W_ih, W_hh, W_comb, hidden, ws);
  attn_logits_kernel<<<BATCH / 256, 256, 0, stream>>>(input, hidden, cell,
                                                      W_attn, b_attn, out_w);
  attn_apply_kernel<<<BATCH / 64, 256, 0, stream>>>(enc, out_w, A16);
  combine_kernel<<<dim3(BATCH / 64, 16), 256, 64 * 65 * sizeof(uint4), stream>>>(
      A16, Wc16, W_comb, b_comb, input, X16);
  gates_lstm_kernel<<<dim3(BATCH / 64, 16), 256, 2 * 64 * 65 * sizeof(uint4), stream>>>(
      X16, H16, Wih16, Whh16, b_ih, b_hh, cell, out_h, out_c);
  out_proj_kernel<<<(BATCH * 32) / 256, 256, 0, stream>>>(out_h, W_out, b_out, out);
}
